// RCLayer_softmax_2010044694578
// MI455X (gfx1250) — matrix operand order repaired and checked by reading, not compiled, not run on hardware
//
#include <hip/hip_runtime.h>

typedef __attribute__((ext_vector_type(16))) _Float16 v16h;
typedef __attribute__((ext_vector_type(8)))  _Float16 v8h;
typedef __attribute__((ext_vector_type(8)))  float    v8f;

#define N_ROWS 32768
#define K_COLS 2048
#define C_DIM  512
#define BM     32          // rows per workgroup (2 m-tiles)
#define THREADS 512        // 16 waves (wave32)
#define NT     8           // n-tiles per wave: 16 waves * 8 = 128 tiles = K 2048
#define KCH    (C_DIM / 32)        // 16 k-chunks

union V16H { v16h v; v8h h[2]; };

// ---------------------------------------------------------------------------
// Pass 1: f32 -> f16 conversion + exact f32 row norms (deterministic reduce).
// ---------------------------------------------------------------------------
__global__ __launch_bounds__(256)
void prep_kernel(const float* __restrict__ inputs,
                 const float* __restrict__ clusters,
                 _Float16* __restrict__ A16,   // may be null (ws too small)
                 _Float16* __restrict__ B16,
                 float* __restrict__ x2,
                 float* __restrict__ c2)
{
    const int row = blockIdx.x;
    const float* src;
    _Float16* dst;
    float* nrm;
    if (row < N_ROWS) {
        src = inputs + (size_t)row * C_DIM;
        dst = A16 ? (A16 + (size_t)row * C_DIM) : nullptr;
        nrm = x2 + row;
    } else {
        const int r = row - N_ROWS;
        src = clusters + (size_t)r * C_DIM;
        dst = B16 + (size_t)r * C_DIM;
        nrm = c2 + r;
    }

    float s = 0.0f;
    for (int i = threadIdx.x; i < C_DIM; i += 256) {
        const float v = src[i];
        s += v * v;
        if (dst) dst[i] = (_Float16)v;
    }
    #pragma unroll
    for (int off = 1; off < 32; off <<= 1) s += __shfl_xor(s, off, 32);

    __shared__ float partial[8];
    if ((threadIdx.x & 31) == 0) partial[threadIdx.x >> 5] = s;
    __syncthreads();
    if (threadIdx.x == 0) {
        float tot = 0.0f;
        #pragma unroll
        for (int i = 0; i < 8; ++i) tot += partial[i];
        *nrm = tot;
    }
}

// ---------------------------------------------------------------------------
// Pass 2: fused WMMA GEMM + d2 + q + row softmax.
// Wave ng (0..15): BOTH m-tiles, n-tiles ng*8+t -> each B fragment feeds
// 2 wmmas; the WG streams B exactly once.
// Full-chunk register ring (8 fragments): during chunk kc, slot t is refilled
// with chunk kc+1 tile t (one uniform immediate offset). Chunk 0 peeled with
// inline C=0; last chunk peeled without refills. kc loop forced rolled.
// ---------------------------------------------------------------------------
template<int USE_A16>
__global__ __launch_bounds__(THREADS)
void fused_wmma_softmax(const float* __restrict__ inputsF32,
                        const _Float16* __restrict__ A16,   // null if !USE_A16
                        const _Float16* __restrict__ B16,
                        const float* __restrict__ x2,
                        const float* __restrict__ c2,
                        float* __restrict__ out)
{
    __shared__ float c2s[K_COLS];
    __shared__ float x2s[BM];
    __shared__ float sumbuf[BM][16];
    __shared__ float rowinv[BM];

    const int tid  = threadIdx.x;
    const int ng   = tid >> 5;      // wave id 0..15
    const int lane = tid & 31;
    const int l15  = lane & 15;
    const int h    = lane >> 4;     // half-wave

    for (int i = tid; i < K_COLS; i += THREADS) c2s[i] = c2[i];
    if (tid < BM) x2s[tid] = x2[blockIdx.x * BM + tid];
    __syncthreads();

    // per-lane bases (elements)
    const size_t arow0 = (size_t)(blockIdx.x * BM + l15) * C_DIM;      // m-tile 0
    const size_t brow  = (size_t)(ng * NT * 16 + l15) * C_DIM;         // t=0 column

    auto loadA = [&](int kc, V16H& a0, V16H& a1) {
        if constexpr (USE_A16) {
            const _Float16* p0 = A16 + arow0 + kc * 32;
            a0.h[0] = *(const v8h*)(p0 + h * 8);
            a0.h[1] = *(const v8h*)(p0 + 16 + h * 8);
            const _Float16* p1 = p0 + 16 * C_DIM;
            a1.h[0] = *(const v8h*)(p1 + h * 8);
            a1.h[1] = *(const v8h*)(p1 + 16 + h * 8);
        } else {
            const float* p0 = inputsF32 + arow0 + kc * 32;
            const v8f f00 = *(const v8f*)(p0 + h * 8);
            const v8f f01 = *(const v8f*)(p0 + 16 + h * 8);
            const float* p1 = p0 + 16 * C_DIM;
            const v8f f10 = *(const v8f*)(p1 + h * 8);
            const v8f f11 = *(const v8f*)(p1 + 16 + h * 8);
            #pragma unroll
            for (int i = 0; i < 8; ++i) {
                a0.h[0][i] = (_Float16)f00[i];
                a0.h[1][i] = (_Float16)f01[i];
                a1.h[0][i] = (_Float16)f10[i];
                a1.h[1][i] = (_Float16)f11[i];
            }
        }
    };

    // per-lane B pointer for the current k-chunk (advances 32 elems = 64B/chunk)
    const _Float16* bpc = B16 + brow + h * 8;

    v8f zero;
    #pragma unroll
    for (int i = 0; i < 8; ++i) zero[i] = 0.0f;

    V16H bb[NT], ac0, ac1, an0, an1;
    v8f acc[2][NT];

    loadA(0, ac0, ac1);
    #pragma unroll
    for (int t = 0; t < NT; ++t) {          // preload whole chunk 0
        const _Float16* rp = bpc + (size_t)t * 16 * C_DIM;
        bb[t].h[0] = *(const v8h*)(rp);
        bb[t].h[1] = *(const v8h*)(rp + 16);
    }

    // ---- chunk 0 peeled: C = 0 (compiler folds to inline constant)
    loadA(1, an0, an1);
    #pragma unroll
    for (int t = 0; t < NT; ++t) {
        acc[0][t] = __builtin_amdgcn_wmma_f32_16x16x32_f16(
            false, ac0.v, false, bb[t].v, (short)0, zero, false, false);
        acc[1][t] = __builtin_amdgcn_wmma_f32_16x16x32_f16(
            false, ac1.v, false, bb[t].v, (short)0, zero, false, false);
        const _Float16* rp = bpc + 32 + (size_t)t * 16 * C_DIM;   // chunk 1, tile t
        bb[t].h[0] = *(const v8h*)(rp);
        bb[t].h[1] = *(const v8h*)(rp + 16);
    }
    bpc += 32;
    ac0 = an0; ac1 = an1;

    // ---- steady state: chunks 1..KCH-2 (rolled)
    #pragma unroll 1
    for (int kc = 1; kc < KCH - 1; ++kc) {
        loadA(kc + 1, an0, an1);            // A prefetch, one chunk ahead
        #pragma unroll
        for (int t = 0; t < NT; ++t) {
            acc[0][t] = __builtin_amdgcn_wmma_f32_16x16x32_f16(
                false, ac0.v, false, bb[t].v, (short)0, acc[0][t], false, false);
            acc[1][t] = __builtin_amdgcn_wmma_f32_16x16x32_f16(
                false, ac1.v, false, bb[t].v, (short)0, acc[1][t], false, false);
            const _Float16* rp = bpc + 32 + (size_t)t * 16 * C_DIM; // next chunk
            bb[t].h[0] = *(const v8h*)(rp);
            bb[t].h[1] = *(const v8h*)(rp + 16);
        }
        bpc += 32;
        ac0 = an0; ac1 = an1;
    }

    // ---- final chunk: consume ring, no refills
    #pragma unroll
    for (int t = 0; t < NT; ++t) {
        acc[0][t] = __builtin_amdgcn_wmma_f32_16x16x32_f16(
            false, ac0.v, false, bb[t].v, (short)0, acc[0][t], false, false);
        acc[1][t] = __builtin_amdgcn_wmma_f32_16x16x32_f16(
            false, ac1.v, false, bb[t].v, (short)0, acc[1][t], false, false);
    }

    // ---- fused softmax: q in (0,1] so exp never overflows -> no max-shift.
    // C/D layout: VGPR r, lane l -> M = m*16 + r + 8*(l>>4), N = l&15.
    float ps[2][8];
    #pragma unroll
    for (int m = 0; m < 2; ++m)
        #pragma unroll
        for (int r = 0; r < 8; ++r) ps[m][r] = 0.0f;

    #pragma unroll
    for (int t = 0; t < NT; ++t) {
        const int crow = (ng * NT + t) * 16 + l15;
        const float c2v = c2s[crow];
        #pragma unroll
        for (int m = 0; m < 2; ++m) {
            #pragma unroll
            for (int r = 0; r < 8; ++r) {
                const int M = m * 16 + h * 8 + r;
                float d2 = x2s[M] + c2v - 2.0f * acc[m][t][r];
                d2 = fmaxf(d2, 0.0f);
                const float q = 1.0f / (1.0f + d2);
                const float e = __expf(q);
                acc[m][t][r] = e;
                ps[m][r] += e;
            }
        }
    }
    #pragma unroll
    for (int off = 1; off < 16; off <<= 1) {
        #pragma unroll
        for (int m = 0; m < 2; ++m)
            #pragma unroll
            for (int r = 0; r < 8; ++r) ps[m][r] += __shfl_xor(ps[m][r], off, 32);
    }
    if (l15 == 0) {
        #pragma unroll
        for (int m = 0; m < 2; ++m)
            #pragma unroll
            for (int r = 0; r < 8; ++r)
                sumbuf[m * 16 + h * 8 + r][ng] = ps[m][r];
    }
    __syncthreads();
    if (tid < BM) {
        float s = 0.0f;
        #pragma unroll
        for (int g = 0; g < 16; ++g) s += sumbuf[tid][g];
        rowinv[tid] = 1.0f / s;
    }
    __syncthreads();

    #pragma unroll
    for (int t = 0; t < NT; ++t) {
        const int crow = (ng * NT + t) * 16 + l15;
        #pragma unroll
        for (int m = 0; m < 2; ++m) {
            #pragma unroll
            for (int r = 0; r < 8; ++r) {
                const int M = m * 16 + h * 8 + r;
                out[(size_t)(blockIdx.x * BM + M) * K_COLS + crow] =
                    acc[m][t][r] * rowinv[M];
            }
        }
    }
}

// ---------------------------------------------------------------------------
extern "C" void kernel_launch(void* const* d_in, const int* in_sizes, int n_in,
                              void* d_out, int out_size, void* d_ws, size_t ws_size,
                              hipStream_t stream) {
    (void)in_sizes; (void)n_in; (void)out_size;
    const float* inputs   = (const float*)d_in[0];
    const float* clusters = (const float*)d_in[1];
    float* out = (float*)d_out;

    // ws layout: [x2 f32 N][c2 f32 K][B16 f16 K*C][A16 f16 N*C (optional)]
    char* ws = (char*)d_ws;
    float*    x2  = (float*)ws;
    float*    c2  = (float*)(ws + (size_t)N_ROWS * sizeof(float));
    _Float16* B16 = (_Float16*)(ws + (size_t)N_ROWS * sizeof(float)
                                   + (size_t)K_COLS * sizeof(float));
    const size_t base = (size_t)N_ROWS * 4 + (size_t)K_COLS * 4
                      + (size_t)K_COLS * C_DIM * 2;
    const int use_a16 = (ws_size >= base + (size_t)N_ROWS * C_DIM * 2) ? 1 : 0;
    _Float16* A16 = use_a16 ? (_Float16*)(ws + base) : nullptr;

    prep_kernel<<<N_ROWS + K_COLS, 256, 0, stream>>>(inputs, clusters,
                                                     A16, B16, x2, c2);
    if (use_a16) {
        fused_wmma_softmax<1><<<N_ROWS / BM, THREADS, 0, stream>>>(
            inputs, A16, B16, x2, c2, out);
    } else {
        fused_wmma_softmax<0><<<N_ROWS / BM, THREADS, 0, stream>>>(
            inputs, nullptr, B16, x2, c2, out);
    }
}
